// TimeIndexedSSM_9345848836687
// MI455X (gfx1250) — matrix operand order repaired and checked by reading, not compiled, not run on hardware
//
#include <hip/hip_runtime.h>
#include <hip/hip_bf16.h>

#define SIN  128
#define TEMB 512
#define EMB  1024
#define SDIM 64
#define BATCH 4
#define TLEN 4096
#define CHUNK 64
#define NCHUNK (TLEN / CHUNK)   // 64
#define UPITCH 68
#define BPITCH 1040             // LDS row pitch for staged B (elements), 2080B, 32B aligned

typedef __attribute__((ext_vector_type(16))) __bf16 v16bf;
typedef __attribute__((ext_vector_type(8)))  float  v8f;

__device__ __forceinline__ float sigmoidf_(float x) { return 1.f / (1.f + __expf(-x)); }
__device__ __forceinline__ float softplusf_(float x) { return (x > 20.f) ? x : log1pf(__expf(x)); }

__device__ __forceinline__ v8f wmma_bf16(v16bf a, v16bf b, v8f c) {
    return __builtin_amdgcn_wmma_f32_16x16x32_bf16(false, a, false, b, (short)0, c, false, false);
}

// ---------------- K0: time-MLP + small params (1 block, 512 threads) -------
__global__ void k0_params(const float* __restrict__ te_g,
                          const float* __restrict__ W1, const float* __restrict__ b1,
                          const float* __restrict__ W2, const float* __restrict__ b2,
                          const float* __restrict__ WA, const float* __restrict__ bA,
                          const float* __restrict__ Wdl, const float* __restrict__ bdl,
                          const float* __restrict__ WD, const float* __restrict__ bD,
                          const float* __restrict__ MA, const float* __restrict__ mA,
                          const float* __restrict__ Mdl, const float* __restrict__ mdl,
                          const float* __restrict__ MD, const float* __restrict__ mD,
                          float* __restrict__ t_out, float* __restrict__ delta_out,
                          float* __restrict__ lam_out, float* __restrict__ D_out) {
    __shared__ float te[SIN];
    __shared__ float h1[TEMB];
    __shared__ float tt[TEMB];
    const int tid = threadIdx.x;  // 512 threads

    if (tid < SIN) te[tid] = te_g[tid];
    __syncthreads();

    // h1 = silu(te @ W1 + b1)
    float acc = b1[tid];
    #pragma unroll 8
    for (int j = 0; j < SIN; ++j) acc = fmaf(te[j], W1[j * TEMB + tid], acc);
    h1[tid] = acc * sigmoidf_(acc);
    __syncthreads();

    // t = h1 @ W2 + b2
    acc = b2[tid];
    #pragma unroll 8
    for (int j = 0; j < TEMB; ++j) acc = fmaf(h1[j], W2[j * TEMB + tid], acc);
    tt[tid] = acc;
    t_out[tid] = acc;
    __syncthreads();

    // A, delta, lam = delta*A   (64 channels)
    if (tid < SDIM) {
        float a = bA[tid], dl = bdl[tid], ga = mA[tid], gd = mdl[tid];
        #pragma unroll 8
        for (int j = 0; j < TEMB; ++j) {
            a  = fmaf(tt[j], WA [j * SDIM + tid], a);
            dl = fmaf(tt[j], Wdl[j * SDIM + tid], dl);
        }
        #pragma unroll 8
        for (int j = 0; j < SIN; ++j) {
            ga = fmaf(te[j], MA [j * SDIM + tid], ga);
            gd = fmaf(te[j], Mdl[j * SDIM + tid], gd);
        }
        float A     = -softplusf_(a) * sigmoidf_(ga);
        float delta = (softplusf_(dl) + 1e-4f) * sigmoidf_(gd);
        delta_out[tid] = delta;
        lam_out[tid]   = delta * A;
    }

    // D vector (1024)
    for (int e = tid; e < EMB; e += 512) {
        float d = bD[e], g = mD[e];
        #pragma unroll 8
        for (int j = 0; j < TEMB; ++j) d = fmaf(tt[j], WD[j * EMB + e], d);
        #pragma unroll 8
        for (int j = 0; j < SIN; ++j) g = fmaf(te[j], MD[j * EMB + e], g);
        D_out[e] = d * sigmoidf_(g);
    }
}

// ---------------- K1: stream WB/MB -> B_bar(bf16 [S,E]), WC/MC -> C(bf16 [E,S])
__global__ void k1_gen_BC(const float* __restrict__ t_ws, const float* __restrict__ te_g,
                          const float* __restrict__ WB, const float* __restrict__ bB,
                          const float* __restrict__ MB, const float* __restrict__ mB,
                          const float* __restrict__ WC, const float* __restrict__ bC,
                          const float* __restrict__ MC, const float* __restrict__ mC,
                          const float* __restrict__ delta,
                          __bf16* __restrict__ Bbf, __bf16* __restrict__ Cbf) {
    __shared__ float ts[TEMB];
    __shared__ float tes[SIN];
    const int tid = threadIdx.x;  // 256
    for (int i = tid; i < TEMB; i += 256) ts[i] = t_ws[i];
    if (tid < SIN) tes[tid] = te_g[tid];
    __syncthreads();

    const int gid = blockIdx.x * 256 + tid;
    if (blockIdx.x < 256) {
        // B_bar[s,e] : idx = s*1024 + e, adjacent threads -> adjacent e (coalesced)
        const int idx = gid;
        const int s = idx >> 10;
        float acc = 0.f;
        #pragma unroll 4
        for (int j = 0; j < TEMB; ++j) {
            __builtin_prefetch(&WB[(size_t)(j + 8) * (SDIM * EMB) + idx], 0, 0);
            acc = fmaf(ts[j], WB[(size_t)j * (SDIM * EMB) + idx], acc);
        }
        float g = 0.f;
        #pragma unroll 4
        for (int j = 0; j < SIN; ++j)
            g = fmaf(tes[j], MB[(size_t)j * (SDIM * EMB) + idx], g);
        float bbar = delta[s] * (acc + bB[idx]) * sigmoidf_(g + mB[idx]);
        Bbf[idx] = (__bf16)bbar;
    } else {
        // C[e,s] : idx = e*64 + s
        const int idx = gid - 256 * 256;
        float acc = 0.f;
        #pragma unroll 4
        for (int j = 0; j < TEMB; ++j) {
            __builtin_prefetch(&WC[(size_t)(j + 8) * (EMB * SDIM) + idx], 0, 0);
            acc = fmaf(ts[j], WC[(size_t)j * (EMB * SDIM) + idx], acc);
        }
        float g = 0.f;
        #pragma unroll 4
        for (int j = 0; j < SIN; ++j)
            g = fmaf(tes[j], MC[(size_t)j * (EMB * SDIM) + idx], g);
        float cv = (acc + bC[idx]) * sigmoidf_(g + mC[idx]);
        Cbf[idx] = (__bf16)cv;
    }
}

// ---- stage B_bar (64 x 1024 bf16) global -> LDS via async copy (ASYNCcnt) --
// LDS layout: row-pitch BPITCH elements (2080B) so WMMA B-fragment reads stay
// 32B-aligned with only 2-way worst-case bank conflicts.
__device__ __forceinline__ void stage_B_lds(__bf16* Bls, const __bf16* __restrict__ Bbf,
                                            int tid) {
    const uint32_t base = (uint32_t)(uintptr_t)Bls;       // LDS byte offset (addr[31:0])
    #pragma unroll 4
    for (int r = 0; r < 32; ++r) {                        // 8192 x 16B chunks / 256 thr
        const int idx  = r * 256 + tid;
        const int row  = idx >> 7;                        // 128 chunks per 2048B row
        const int colb = (idx & 127) << 4;                // byte offset within row
        const uint32_t lds = base + (uint32_t)(row * (BPITCH * 2) + colb);
        const void* g = (const char*)Bbf + row * 2048 + colb;
        asm volatile("global_load_async_to_lds_b128 %0, %1, off"
                     :: "v"(lds), "v"(g) : "memory");
    }
    asm volatile("s_wait_asynccnt 0x0" ::: "memory");     // own wave's asyncs done
}

// ---- shared GEMM1: U[64,64] = Xchunk[64,1024] (f32->bf16) @ B_bar^T --------
// A-fragment: lane l<16 holds row M=l, K runs contiguous {k0+16kh .. k0+16kh+15}
__device__ __forceinline__ v16bf load_a_frag_x(const float* __restrict__ xrow, int k0, int kh) {
    const float4* p0 = (const float4*)(xrow + k0 + 16 * kh);
    const float4* p1 = (const float4*)(xrow + k0 + 16 * kh + 8);
    float4 a = p0[0], b = p0[1], c = p1[0], d = p1[1];
    v16bf r;
    r[0] = (__bf16)a.x;  r[1] = (__bf16)a.y;  r[2] = (__bf16)a.z;  r[3] = (__bf16)a.w;
    r[4] = (__bf16)b.x;  r[5] = (__bf16)b.y;  r[6] = (__bf16)b.z;  r[7] = (__bf16)b.w;
    r[8] = (__bf16)c.x;  r[9] = (__bf16)c.y;  r[10] = (__bf16)c.z; r[11] = (__bf16)c.w;
    r[12] = (__bf16)d.x; r[13] = (__bf16)d.y; r[14] = (__bf16)d.z; r[15] = (__bf16)d.w;
    return r;
}

__device__ __forceinline__ void gemm1_U(const float* __restrict__ xbase,
                                        const __bf16* Bls,   // staged in LDS
                                        float (*Uls)[UPITCH], int wave, int lane) {
    const int mi  = wave >> 1;          // M-tile (time)
    const int ni0 = (wave & 1) * 2;     // first of two N-tiles (state)
    const int l15 = lane & 15;
    const int kh  = lane >> 4;
    const float*  xrow = xbase + (size_t)(mi * 16 + l15) * EMB;
    const __bf16* bp0  = Bls + (size_t)(ni0 * 16 + l15) * BPITCH + kh * 16;
    const __bf16* bp1  = bp0 + (size_t)16 * BPITCH;
    v8f acc0 = {}; v8f acc1 = {};
    #pragma unroll 4
    for (int k0 = 0; k0 < EMB; k0 += 32) {
        v16bf a  = load_a_frag_x(xrow, k0, kh);
        v16bf b0 = *(const v16bf*)(bp0 + k0);   // LDS: 32B, K = k0+kh*16 .. +15
        v16bf b1 = *(const v16bf*)(bp1 + k0);
        acc0 = wmma_bf16(a, b0, acc0);
        acc1 = wmma_bf16(a, b1, acc1);
    }
    const int rbase = mi * 16 + 8 * kh;
    #pragma unroll
    for (int v = 0; v < 8; ++v) {
        Uls[rbase + v][ni0 * 16 + l15]       = acc0[v];
        Uls[rbase + v][(ni0 + 1) * 16 + l15] = acc1[v];
    }
}

// ---------------- K2: pass A — per-chunk decayed sum S_c ---------------------
__global__ void k2_passA(const float* __restrict__ x, const __bf16* __restrict__ Bbf,
                         const float* __restrict__ lam, float* __restrict__ Sc) {
    __shared__ float Uls[CHUNK][UPITCH];
    __shared__ __align__(32) __bf16 Bls[SDIM * BPITCH];
    const int c = blockIdx.x, b = blockIdx.y;
    const int tid = threadIdx.x, wave = tid >> 5, lane = tid & 31;
    stage_B_lds(Bls, Bbf, tid);
    __syncthreads();
    const float* xbase = x + (((size_t)b * TLEN) + (size_t)c * CHUNK) * EMB;
    gemm1_U(xbase, Bls, Uls, wave, lane);
    __syncthreads();
    if (tid < SDIM) {
        const float a = __expf(lam[tid]);
        float h = 0.f;
        #pragma unroll 8
        for (int l = 0; l < CHUNK; ++l) h = fmaf(a, h, Uls[l][tid]);
        Sc[((size_t)c * BATCH + b) * SDIM + tid] = h;   // = sum_l a^(63-l) u_l
    }
}

// ---------------- K3: sequential prefix over chunks (tiny) -------------------
__global__ void k3_prefix(const float* __restrict__ lam, const float* __restrict__ Sc,
                          float* __restrict__ hin) {
    const int tid = threadIdx.x;       // 256 = BATCH * SDIM
    const int b = tid >> 6, s = tid & 63;
    const float a64 = __expf(64.f * lam[s]);
    float h = 0.f;
    for (int c = 0; c < NCHUNK; ++c) {
        const size_t i = ((size_t)c * BATCH + b) * SDIM + s;
        hin[i] = h;
        h = fmaf(a64, h, Sc[i]);
    }
}

// ---------------- K4: pass C — full states + Y = H*C^T + D*x -----------------
__global__ void k4_passC(const float* __restrict__ x, const __bf16* __restrict__ Bbf,
                         const __bf16* __restrict__ Cbf, const float* __restrict__ lam,
                         const float* __restrict__ hin, const float* __restrict__ Dw,
                         float* __restrict__ out) {
    __shared__ float Uls[CHUNK][UPITCH];
    __shared__ __align__(32) __bf16 Bls[SDIM * BPITCH];
    const int c = blockIdx.x, b = blockIdx.y;
    const int tid = threadIdx.x, wave = tid >> 5, lane = tid & 31;
    stage_B_lds(Bls, Bbf, tid);
    __syncthreads();
    const float* xbase = x + (((size_t)b * TLEN) + (size_t)c * CHUNK) * EMB;
    gemm1_U(xbase, Bls, Uls, wave, lane);   // x chunk is L2-resident (67MB < 192MB L2)
    __syncthreads();

    if (tid < SDIM) {
        const float a = __expf(lam[tid]);
        float h = hin[((size_t)c * BATCH + b) * SDIM + tid];
        #pragma unroll 8
        for (int l = 0; l < CHUNK; ++l) {
            h = fmaf(a, h, Uls[l][tid]);
            Uls[l][tid] = h;               // in-place: U -> H
        }
    }
    __syncthreads();

    // GEMM2: Y[64,1024] = H[64,64] @ C^T ;  wave -> fixed M-tile, 32 N-tiles
    const int mi = wave & 3;
    const int ng = wave >> 2;              // N-tile group (0/1)
    const int l15 = lane & 15, kh = lane >> 4;
    const float* hrow = &Uls[mi * 16 + l15][0];
    v16bf a0, a1;                           // A-fragments for k0 = 0 and 32
    #pragma unroll
    for (int e = 0; e < 8; ++e) {
        a0[e]     = (__bf16)hrow[16 * kh + e];
        a0[e + 8] = (__bf16)hrow[16 * kh + 8 + e];
        a1[e]     = (__bf16)hrow[32 + 16 * kh + e];
        a1[e + 8] = (__bf16)hrow[40 + 16 * kh + e];
    }
    for (int j = 0; j < 32; ++j) {
        const int nI = ng * 32 + j;
        const int e  = nI * 16 + l15;
        const __bf16* cp = Cbf + (size_t)e * SDIM + kh * 16;   // C stays in L2
        v16bf b0 = *(const v16bf*)cp;          // K = kh*16 .. +15
        v16bf b1 = *(const v16bf*)(cp + 32);   // K = 32 + kh*16 .. +15
        v8f acc = {};
        acc = wmma_bf16(a0, b0, acc);
        acc = wmma_bf16(a1, b1, acc);
        const float Dv = Dw[e];
        const int tbase = c * CHUNK + mi * 16 + 8 * kh;
        const size_t base = (((size_t)b * TLEN) + tbase) * EMB + e;
        #pragma unroll
        for (int v = 0; v < 8; ++v) {
            const size_t idx = base + (size_t)v * EMB;
            out[idx] = acc[v] + Dv * x[idx];
        }
    }
}

// ---------------------------------------------------------------------------
extern "C" void kernel_launch(void* const* d_in, const int* in_sizes, int n_in,
                              void* d_out, int out_size, void* d_ws, size_t ws_size,
                              hipStream_t stream) {
    const float* te  = (const float*)d_in[0];
    const float* x   = (const float*)d_in[1];
    const float* W1  = (const float*)d_in[2];  const float* b1  = (const float*)d_in[3];
    const float* W2  = (const float*)d_in[4];  const float* b2  = (const float*)d_in[5];
    const float* WA  = (const float*)d_in[6];  const float* bA  = (const float*)d_in[7];
    const float* WB  = (const float*)d_in[8];  const float* bB  = (const float*)d_in[9];
    const float* WC  = (const float*)d_in[10]; const float* bC  = (const float*)d_in[11];
    const float* WD  = (const float*)d_in[12]; const float* bD  = (const float*)d_in[13];
    const float* Wdl = (const float*)d_in[14]; const float* bdl = (const float*)d_in[15];
    const float* MA  = (const float*)d_in[16]; const float* mA  = (const float*)d_in[17];
    const float* MB  = (const float*)d_in[18]; const float* mB  = (const float*)d_in[19];
    const float* MC  = (const float*)d_in[20]; const float* mC  = (const float*)d_in[21];
    const float* MD  = (const float*)d_in[22]; const float* mD  = (const float*)d_in[23];
    const float* Mdl = (const float*)d_in[24]; const float* mdl = (const float*)d_in[25];
    float* out = (float*)d_out;

    // workspace layout (byte offsets, 64B aligned)
    char* ws = (char*)d_ws;
    float*  t_ws   = (float*)(ws + 0);        // 512 f32
    float*  delta  = (float*)(ws + 2048);     // 64
    float*  lamv   = (float*)(ws + 2304);     // 64
    float*  Dv     = (float*)(ws + 2560);     // 1024
    __bf16* Bbf    = (__bf16*)(ws + 6656);    // 64*1024 bf16  [S,E]
    __bf16* Cbf    = (__bf16*)(ws + 137728);  // 1024*64 bf16  [E,S]
    float*  Sc     = (float*)(ws + 268800);   // NCHUNK*BATCH*SDIM
    float*  hin    = (float*)(ws + 334336);   // NCHUNK*BATCH*SDIM

    k0_params<<<1, 512, 0, stream>>>(te, W1, b1, W2, b2, WA, bA, Wdl, bdl, WD, bD,
                                     MA, mA, Mdl, mdl, MD, mD,
                                     t_ws, delta, lamv, Dv);
    k1_gen_BC<<<512, 256, 0, stream>>>(t_ws, te, WB, bB, MB, mB, WC, bC, MC, mC,
                                       delta, Bbf, Cbf);
    dim3 grid(NCHUNK, BATCH);
    k2_passA<<<grid, 256, 0, stream>>>(x, Bbf, lamv, Sc);
    k3_prefix<<<1, 256, 0, stream>>>(lamv, Sc, hin);
    k4_passC<<<grid, 256, 0, stream>>>(x, Bbf, Cbf, lamv, hin, Dv, out);
}
